// TypeAttentiveGCN_19250043421248
// MI455X (gfx1250) — matrix operand order repaired and checked by reading, not compiled, not run on hardware
//
#include <hip/hip_runtime.h>
#include <hip/hip_fp16.h>

// ---------------------------------------------------------------------------
// TypeAttentiveGCN, algebraically factored for MI455X (gfx1250, wave32).
//   N=256 nodes, D=H=768, T=100 dep types.
// All dense GEMMs run on v_wmma_f32_16x16x32_f16 (f16 in, f32 accum).
// ---------------------------------------------------------------------------

#define NN   256
#define DD   768
#define HH   768
#define TT   100
#define TPAD 112   // T padded to multiple of 16 for WMMA M-tiles

typedef __attribute__((ext_vector_type(16))) _Float16     v16h;
typedef __attribute__((ext_vector_type(8)))  float        v8f;
typedef __attribute__((ext_vector_type(8)))  unsigned int v8u;

union Frag16 { v8u u; v16h h; };

// -------------------------- conversion kernels -----------------------------

__global__ void k_f32_to_f16(const float* __restrict__ src,
                             __half* __restrict__ dst, int n) {
  int i = blockIdx.x * blockDim.x + threadIdx.x;
  if (i < n) dst[i] = __float2half(src[i]);
}

__global__ void k_f32_to_f16_pad(const float* __restrict__ src,
                                 __half* __restrict__ dst, int n, int ntot) {
  int i = blockIdx.x * blockDim.x + threadIdx.x;
  if (i < ntot) dst[i] = (i < n) ? __float2half(src[i]) : __float2half(0.0f);
}

// W: [K x N] row-major (f32)  ->  Wt: [N x K] row-major (f16)
__global__ void k_transpose_f32_to_f16(const float* __restrict__ W,
                                       __half* __restrict__ Wt, int K, int N) {
  int idx = blockIdx.x * blockDim.x + threadIdx.x;
  if (idx < K * N) {
    int k = idx / N, n = idx % N;
    Wt[(size_t)n * K + k] = __float2half(W[idx]);
  }
}

// ------------------------------ WMMA GEMM ----------------------------------
// C[M,N] = A[M,K](f16) * Bt[N,K]^T(f16) + bias[N]   (f32 accumulate/output)
// One wave per block; each wave owns a 16x64 output tile (A reused 4x).
__global__ __launch_bounds__(32)
void k_wmma_gemm(const __half* __restrict__ A, const __half* __restrict__ Bt,
                 const float* __restrict__ bias, float* __restrict__ C,
                 int M, int N, int K) {
  const int lane   = threadIdx.x & 31;
  const int lane16 = lane & 15;
  const int half16 = lane >> 4;               // 0: lanes 0-15, 1: lanes 16-31
  const int mTile  = blockIdx.y;              // 16 rows
  const int nBase  = blockIdx.x * 64;         // 64 cols -> 4 WMMA tiles

  const int mRow = mTile * 16 + lane16;       // A-fragment row for this lane

  v8f acc[4] = {};                            // four 16x16 f32 accumulators

  const unsigned* Arow =
      reinterpret_cast<const unsigned*>(A + (size_t)mRow * K);

  for (int k0 = 0; k0 < K; k0 += 32) {
    // --- A fragment (16x32 f16): ISA 7.12.2 layout ---
    Frag16 a;
    const int ka = k0 + half16 * 8;
#pragma unroll
    for (int r = 0; r < 4; ++r) a.u[r]     = Arow[(ka + 2 * r) >> 1];
#pragma unroll
    for (int r = 0; r < 4; ++r) a.u[4 + r] = Arow[(ka + 16 + 2 * r) >> 1];

    // speculative prefetch of next K-step of A (global_prefetch_b8)
    __builtin_prefetch((const void*)(Arow + ((k0 + 32) >> 1)), 0, 0);

    // --- 4 B fragments (32x16 f16 each), A reused ---
#pragma unroll
    for (int t = 0; t < 4; ++t) {
      const int n = nBase + t * 16 + lane16;
      const unsigned* Brow =
          reinterpret_cast<const unsigned*>(Bt + (size_t)n * K);
      Frag16 b;
      const int kb = k0 + half16 * 8;
#pragma unroll
      for (int r = 0; r < 8; ++r) b.u[r] = Brow[(kb + 2 * r + (r >= 4 ? 8 : 0)) >> 1];

      acc[t] = __builtin_amdgcn_wmma_f32_16x16x32_f16(
          /*neg_a=*/false, a.h, /*neg_b=*/false, b.h,
          /*c_mod=*/(short)0, acc[t], /*reuse_a=*/false, /*reuse_b=*/false);
    }
  }

  // --- store D (16x16 f32: VGPR r -> row r + 8*half16) ---
  const int mOut = mTile * 16 + half16 * 8;
#pragma unroll
  for (int t = 0; t < 4; ++t) {
    const int n  = nBase + t * 16 + lane16;
    const float bv = bias ? bias[n] : 0.0f;
#pragma unroll
    for (int r = 0; r < 8; ++r)
      C[(size_t)(mOut + r) * N + n] = acc[t][r] + bv;
  }
}

// --------------------------- pairwise stage --------------------------------
// Per row i: y_ij = Wg_b + Wg . tanh(hi_i + hj_j + embT[t_ij])
//            p    = softmax(y * mask)
//            s_i  = sum_j p_ij * tanh(hE_j + embE[t_ij])   (f16 out for WMMA)
__global__ __launch_bounds__(256)
void k_pairwise(const float* __restrict__ hi, const float* __restrict__ hj,
                const float* __restrict__ embT, const float* __restrict__ hE,
                const float* __restrict__ embE, const int* __restrict__ adj,
                const int* __restrict__ dep, const float* __restrict__ Wg_w,
                const float* __restrict__ Wg_b, __half* __restrict__ s_out) {
  const int i   = blockIdx.x;
  const int tid = threadIdx.x;

  __shared__ float hi_s[HH];
  __shared__ float y_s[NN];
  __shared__ float p_s[NN];
  __shared__ int   t_s[NN];
  __shared__ float red[NN];

#pragma unroll
  for (int c = 0; c < HH / NN; ++c)
    hi_s[tid + NN * c] = hi[(size_t)i * HH + tid + NN * c];
  t_s[tid] = dep[(size_t)i * NN + tid];
  __syncthreads();

  // phase 1: scores (thread j owns one neighbor)
  {
    const int j = tid;
    const float* hjr = hj + (size_t)j * HH;
    const float* er  = embT + (size_t)t_s[j] * HH;
    float acc = 0.0f;
    for (int hh = 0; hh < HH; ++hh)
      acc += Wg_w[hh] * tanhf(hi_s[hh] + hjr[hh] + er[hh]);
    const float yv = acc + Wg_b[0];
    // faithful to reference: scores MULTIPLIED by mask, then softmax
    y_s[j] = (adj[(size_t)i * NN + j] > 0) ? yv : 0.0f;
  }
  __syncthreads();

  // phase 2: softmax over 256 values
  red[tid] = y_s[tid];
  __syncthreads();
  for (int off = 128; off > 0; off >>= 1) {
    if (tid < off) red[tid] = fmaxf(red[tid], red[tid + off]);
    __syncthreads();
  }
  const float mx = red[0];
  __syncthreads();
  const float ev = __expf(y_s[tid] - mx);
  red[tid] = ev;
  __syncthreads();
  for (int off = 128; off > 0; off >>= 1) {
    if (tid < off) red[tid] += red[tid + off];
    __syncthreads();
  }
  p_s[tid] = ev / red[0];
  __syncthreads();

  // phase 3: weighted tanh reduction (thread owns 3 H-channels, coalesced)
#pragma unroll
  for (int c = 0; c < HH / NN; ++c) {
    const int hh = tid + NN * c;
    float acc = 0.0f;
    for (int j = 0; j < NN; ++j)
      acc += p_s[j] * tanhf(hE[(size_t)j * HH + hh] +
                            embE[(size_t)t_s[j] * HH + hh]);
    s_out[(size_t)i * HH + hh] = __float2half(acc);
  }
}

// ----------------------- residual + LayerNorm + ReLU -----------------------
__global__ __launch_bounds__(256)
void k_add_ln_relu(const float* __restrict__ xbuf, const float* __restrict__ hr,
                   const float* __restrict__ gamma, const float* __restrict__ beta,
                   float* __restrict__ out) {
  const int i   = blockIdx.x;
  const int tid = threadIdx.x;
  __shared__ float red[NN];

  float x[HH / NN];
  float ssum = 0.0f;
#pragma unroll
  for (int c = 0; c < HH / NN; ++c) {
    const int hh = tid + NN * c;
    x[c] = xbuf[(size_t)i * HH + hh] + hr[(size_t)i * HH + hh];
    ssum += x[c];
  }
  red[tid] = ssum;
  __syncthreads();
  for (int off = 128; off > 0; off >>= 1) {
    if (tid < off) red[tid] += red[tid + off];
    __syncthreads();
  }
  const float mu = red[0] * (1.0f / HH);
  __syncthreads();

  float ssq = 0.0f;
#pragma unroll
  for (int c = 0; c < HH / NN; ++c) { const float d = x[c] - mu; ssq += d * d; }
  red[tid] = ssq;
  __syncthreads();
  for (int off = 128; off > 0; off >>= 1) {
    if (tid < off) red[tid] += red[tid + off];
    __syncthreads();
  }
  const float rstd = rsqrtf(red[0] * (1.0f / HH) + 1e-5f);

#pragma unroll
  for (int c = 0; c < HH / NN; ++c) {
    const int hh = tid + NN * c;
    const float v = (x[c] - mu) * rstd * gamma[hh] + beta[hh];
    out[(size_t)i * HH + hh] = fmaxf(v, 0.0f);
  }
}

// ------------------------------ launcher -----------------------------------

extern "C" void kernel_launch(void* const* d_in, const int* in_sizes, int n_in,
                              void* d_out, int out_size, void* d_ws, size_t ws_size,
                              hipStream_t stream) {
  const float* h     = (const float*)d_in[0];
  const int*   adj   = (const int*)  d_in[1];
  const int*   dep   = (const int*)  d_in[2];
  const float* emb   = (const float*)d_in[3];
  const float* Wi_w  = (const float*)d_in[4];
  const float* Wi_b  = (const float*)d_in[5];
  const float* Wj_w  = (const float*)d_in[6];
  const float* Wj_b  = (const float*)d_in[7];
  const float* Wlp_w = (const float*)d_in[8];
  const float* Wlp_b = (const float*)d_in[9];
  const float* Wg_w  = (const float*)d_in[10];
  const float* Wg_b  = (const float*)d_in[11];
  const float* We_w  = (const float*)d_in[12];  // [2D, H]
  const float* We_b  = (const float*)d_in[13];
  const float* Wk_w  = (const float*)d_in[14];
  const float* Wk_b  = (const float*)d_in[15];
  const float* Wr_w  = (const float*)d_in[16];
  const float* Wr_b  = (const float*)d_in[17];
  const float* gamma = (const float*)d_in[18];
  const float* beta  = (const float*)d_in[19];

  char*  ws  = (char*)d_ws;
  size_t off = 0;
  auto alloc = [&](size_t bytes) -> void* {
    void* p = ws + off;
    off = (off + bytes + 255) & ~(size_t)255;
    return p;
  };

  __half* h_f16   = (__half*)alloc((size_t)NN * DD * 2);
  __half* emb_f16 = (__half*)alloc((size_t)TPAD * DD * 2);
  __half* Wt_i    = (__half*)alloc((size_t)DD * HH * 2);
  __half* Wt_j    = (__half*)alloc((size_t)DD * HH * 2);
  __half* Wt_lp   = (__half*)alloc((size_t)DD * HH * 2);
  __half* Wt_elo  = (__half*)alloc((size_t)DD * HH * 2);
  __half* Wt_ehi  = (__half*)alloc((size_t)DD * HH * 2);
  __half* Wt_k    = (__half*)alloc((size_t)HH * HH * 2);
  __half* Wt_r    = (__half*)alloc((size_t)DD * HH * 2);
  float*  hi      = (float*)alloc((size_t)NN * HH * 4);
  float*  hj      = (float*)alloc((size_t)NN * HH * 4);
  float*  hE      = (float*)alloc((size_t)NN * HH * 4);
  float*  hr      = (float*)alloc((size_t)NN * HH * 4);
  float*  embT    = (float*)alloc((size_t)TPAD * HH * 4);
  float*  embE    = (float*)alloc((size_t)TPAD * HH * 4);
  __half* s_f16   = (__half*)alloc((size_t)NN * HH * 2);
  float*  xbuf    = (float*)alloc((size_t)NN * HH * 4);
  (void)ws_size; (void)in_sizes; (void)n_in; (void)out_size;

  // --- conversions / transposes ---
  {
    int n = NN * DD;
    k_f32_to_f16<<<(n + 255) / 256, 256, 0, stream>>>(h, h_f16, n);
  }
  {
    int n = TT * DD, ntot = TPAD * DD;
    k_f32_to_f16_pad<<<(ntot + 255) / 256, 256, 0, stream>>>(emb, emb_f16, n, ntot);
  }
  {
    int n = DD * HH;
    int g = (n + 255) / 256;
    k_transpose_f32_to_f16<<<g, 256, 0, stream>>>(Wi_w,             Wt_i,   DD, HH);
    k_transpose_f32_to_f16<<<g, 256, 0, stream>>>(Wj_w,             Wt_j,   DD, HH);
    k_transpose_f32_to_f16<<<g, 256, 0, stream>>>(Wlp_w,            Wt_lp,  DD, HH);
    k_transpose_f32_to_f16<<<g, 256, 0, stream>>>(We_w,             Wt_elo, DD, HH);
    k_transpose_f32_to_f16<<<g, 256, 0, stream>>>(We_w + (size_t)DD * HH, Wt_ehi, DD, HH);
    k_transpose_f32_to_f16<<<g, 256, 0, stream>>>(Wk_w,             Wt_k,   HH, HH);
    k_transpose_f32_to_f16<<<g, 256, 0, stream>>>(Wr_w,             Wt_r,   DD, HH);
  }

  // --- WMMA GEMMs: node-side projections + type-embedding projections ---
  dim3 blk(32, 1, 1);
  dim3 gN(HH / 64, NN / 16, 1);    // (12, 16)
  dim3 gT(HH / 64, TPAD / 16, 1);  // (12, 7)
  k_wmma_gemm<<<gN, blk, 0, stream>>>(h_f16,   Wt_i,   Wi_b,    hi,   NN,   HH, DD);
  k_wmma_gemm<<<gN, blk, 0, stream>>>(h_f16,   Wt_j,   Wj_b,    hj,   NN,   HH, DD);
  k_wmma_gemm<<<gN, blk, 0, stream>>>(h_f16,   Wt_elo, nullptr, hE,   NN,   HH, DD);
  k_wmma_gemm<<<gN, blk, 0, stream>>>(h_f16,   Wt_r,   Wr_b,    hr,   NN,   HH, DD);
  k_wmma_gemm<<<gT, blk, 0, stream>>>(emb_f16, Wt_lp,  Wlp_b,   embT, TPAD, HH, DD);
  k_wmma_gemm<<<gT, blk, 0, stream>>>(emb_f16, Wt_ehi, We_b,    embE, TPAD, HH, DD);

  // --- pairwise attention + weighted tanh reduction ---
  k_pairwise<<<NN, 256, 0, stream>>>(hi, hj, embT, hE, embE, adj, dep,
                                     Wg_w, Wg_b, s_f16);

  // --- x = s @ Wk + Wk_b (WMMA), then +residual, LayerNorm, ReLU ---
  k_wmma_gemm<<<gN, blk, 0, stream>>>(s_f16, Wt_k, Wk_b, xbuf, NN, HH, HH);
  k_add_ln_relu<<<NN, 256, 0, stream>>>(xbuf, hr, gamma, beta, (float*)d_out);
}
